// GPT_13091060318987
// MI455X (gfx1250) — matrix operand order repaired and checked by reading, not compiled, not run on hardware
//
#include <hip/hip_runtime.h>
#include <hip/hip_bf16.h>

typedef __bf16 bf16_t;
typedef __attribute__((ext_vector_type(16))) __bf16 v16bf;
typedef __attribute__((ext_vector_type(8)))  __bf16 bf16x8;
typedef __attribute__((ext_vector_type(8)))  float  v8f;
typedef __attribute__((ext_vector_type(4)))  int    v4i;

#define LL  6
#define HH  16
#define CC  1024
#define HDD 64
#define VV  32000
#define TT  1024
#define BBATCH 2
#define BT  (BBATCH*TT)
#define DFF (4*CC)

// ---- gfx1250 async global->LDS staging (ASYNCcnt path), guarded ----------
#if defined(__has_builtin)
#if __has_builtin(__builtin_amdgcn_global_load_async_to_lds_b128)
#define USE_ASYNC_LDS 1
#endif
#endif

typedef __attribute__((address_space(1))) v4i* gv4i_p;   // global int4*
typedef __attribute__((address_space(3))) v4i* lv4i_p;   // LDS int4*

__device__ __forceinline__ void copy16_g2l(bf16_t* ldst, const bf16_t* gsrc) {
#ifdef USE_ASYNC_LDS
  __builtin_amdgcn_global_load_async_to_lds_b128(
      (gv4i_p)gsrc, (lv4i_p)ldst, 0, 0);
#else
  *(bf16x8*)ldst = *(const bf16x8*)gsrc;
#endif
}

__device__ __forceinline__ void wait_async_all() {
#ifdef USE_ASYNC_LDS
#if __has_builtin(__builtin_amdgcn_s_wait_asynccnt)
  __builtin_amdgcn_s_wait_asynccnt(0);
#else
  asm volatile("s_wait_asynccnt 0x0" ::: "memory");
#endif
#endif
}

__device__ __forceinline__ bf16_t to_bf16(float f) {
  unsigned int u = __float_as_uint(f);
  u += 0x7FFFu + ((u >> 16) & 1u);               // round-to-nearest-even
  unsigned short hs = (unsigned short)(u >> 16);
  bf16_t r;
  __builtin_memcpy(&r, &hs, 2);
  return r;
}

__device__ __forceinline__ v16bf combine8(const bf16_t* p0, const bf16_t* p1) {
  bf16x8 a = *(const bf16x8*)p0;
  bf16x8 b = *(const bf16x8*)p1;
  v16bf r;
#pragma unroll
  for (int i = 0; i < 8; ++i) { r[i] = a[i]; r[i + 8] = b[i]; }
  return r;
}

// ---------------------------------------------------------------------------
// f32 -> bf16 bulk convert (weights)
// ---------------------------------------------------------------------------
__global__ void cvt_kernel(const float* __restrict__ in, bf16_t* __restrict__ out, size_t n) {
  size_t i = (size_t)blockIdx.x * blockDim.x + threadIdx.x;
  size_t stride = (size_t)gridDim.x * blockDim.x;
  for (; i < n; i += stride) out[i] = to_bf16(in[i]);
}

// ---------------------------------------------------------------------------
// token + position embedding -> fp32 residual stream
// ---------------------------------------------------------------------------
__global__ void embed_kernel(const int* __restrict__ idx, const float* __restrict__ wte,
                             const float* __restrict__ wpe, float* __restrict__ x) {
  int i = blockIdx.x * blockDim.x + threadIdx.x;
  if (i >= BT * CC) return;
  int bt = i / CC, c = i - bt * CC;
  int tok = idx[bt];
  x[i] = wte[(size_t)tok * CC + c] + wpe[(size_t)(bt % TT) * CC + c];
}

// ---------------------------------------------------------------------------
// LayerNorm: fp32 in -> bf16 out, one block (8 waves) per row of C=1024
// ---------------------------------------------------------------------------
__global__ __launch_bounds__(256)
void ln_kernel(const float* __restrict__ x, const float* __restrict__ g,
               const float* __restrict__ bb, bf16_t* __restrict__ out) {
  const int row = blockIdx.x;
  const float* xr = x + (size_t)row * CC;
  float s = 0.f, s2 = 0.f;
  for (int c = threadIdx.x; c < CC; c += 256) { float v = xr[c]; s += v; s2 += v * v; }
  for (int msk = 1; msk < 32; msk <<= 1) {
    s += __shfl_xor(s, msk, 32);
    s2 += __shfl_xor(s2, msk, 32);
  }
  __shared__ float red[2][8];
  int wid = threadIdx.x >> 5, lane = threadIdx.x & 31;
  if (lane == 0) { red[0][wid] = s; red[1][wid] = s2; }
  __syncthreads();
  float ts = 0.f, ts2 = 0.f;
#pragma unroll
  for (int i = 0; i < 8; ++i) { ts += red[0][i]; ts2 += red[1][i]; }
  float mean = ts * (1.f / CC);
  float var  = ts2 * (1.f / CC) - mean * mean;
  float rstd = rsqrtf(var + 1e-5f);
  for (int c = threadIdx.x; c < CC; c += 256)
    out[(size_t)row * CC + c] = to_bf16((xr[c] - mean) * rstd * g[c] + bb[c]);
}

// ---------------------------------------------------------------------------
// bf16 WMMA GEMM: out[M,N] = A[M,K] * B + epilogue
//   B_NK == 0 : B stored [K,N] row-major (qkv/aproj/fc/mproj weights)
//   B_NK == 1 : B stored [N,K] row-major (tied lm_head uses wte[V,C])
// Block: 256 thr = 8 waves, block tile 128x64, wave tile 32x32 (4 wmma accums)
// A-tile (and B-tile when k-contiguous) staged via GLOBAL_LOAD_ASYNC_TO_LDS.
// ---------------------------------------------------------------------------
template <int B_NK>
__global__ __launch_bounds__(256)
void gemm_kernel(const bf16_t* __restrict__ A, const bf16_t* __restrict__ B,
                 int M, int N, int K,
                 const float* __restrict__ bias, int do_gelu,
                 const float* __restrict__ resid,
                 bf16_t* __restrict__ outb, float* __restrict__ outf) {
  (void)M;
  const int tid  = threadIdx.x;
  const int lane = tid & 31;
  const int wid  = tid >> 5;
  const int lm   = lane & 15;
  const int hf   = lane >> 4;
  const int m0 = blockIdx.y * 128;
  const int n0 = blockIdx.x * 64;
  const int wm = (wid >> 1) * 32;   // 0,32,64,96
  const int wn = (wid & 1) * 32;    // 0,32

  __shared__ bf16_t As[128 * 40];   // stride 40 elems = 80B: 16B-aligned, conflict-free
  __shared__ bf16_t Bs[64 * 40];    // stored [n][k]

  v8f acc[2][2] = {};

  // per-thread staging coordinates (loop-invariant)
  const int arow = tid >> 1;
  const int aco  = (tid & 1) * 16;
  const bf16_t* gA = A + (size_t)(m0 + arow) * K + aco;

  for (int k0 = 0; k0 < K; k0 += 32) {
    {   // stage A tile 128x32 (row-major, contiguous k) -> async to LDS
      const bf16_t* gp = gA + k0;
      copy16_g2l(&As[arow * 40 + aco],     gp);
      copy16_g2l(&As[arow * 40 + aco + 8], gp + 8);
      if (k0 + 32 < K) __builtin_prefetch(gp + 32, 0, 1);   // next K-tile of A
    }
    if (B_NK) {   // B [N,K]: k-contiguous per n row -> async to LDS
      int n  = tid >> 2;
      int co = (tid & 3) * 8;
      const bf16_t* gp = B + (size_t)(n0 + n) * K + k0 + co;
      copy16_g2l(&Bs[n * 40 + co], gp);
      if (k0 + 32 < K) __builtin_prefetch(gp + 32, 0, 1);
    } else {      // B [K,N]: read rows of N, scatter-transpose into Bs[n][k]
      int kk = tid & 31;
      int n8 = (tid >> 5) * 8;
      const bf16_t* gp = B + (size_t)(k0 + kk) * N + n0 + n8;
      bf16x8 v = *(const bf16x8*)gp;
#pragma unroll
      for (int i = 0; i < 8; ++i) Bs[(n8 + i) * 40 + kk] = v[i];
      if (k0 + 32 < K) __builtin_prefetch(gp + (size_t)32 * N, 0, 1);  // next K-tile of B
    }
    wait_async_all();
    __syncthreads();

    v16bf af[2], bfr[2];
#pragma unroll
    for (int mt = 0; mt < 2; ++mt) {   // A frag 16x32: lane=m, K runs {8*hf..}, {16+8*hf..}
      const bf16_t* p = &As[(wm + mt * 16 + lm) * 40];
      af[mt] = combine8(p + 8 * hf, p + 16 + 8 * hf);
    }
#pragma unroll
    for (int nt = 0; nt < 2; ++nt) {   // B frag 32x16: lane=n, K runs {8*hf..}, {16+8*hf..}
      const bf16_t* p = &Bs[(wn + nt * 16 + lm) * 40];
      bfr[nt] = combine8(p + 8 * hf, p + 16 + 8 * hf);
    }
#pragma unroll
    for (int mt = 0; mt < 2; ++mt)
#pragma unroll
      for (int nt = 0; nt < 2; ++nt)
        acc[mt][nt] = __builtin_amdgcn_wmma_f32_16x16x32_bf16(
            false, af[mt], false, bfr[nt], (short)0, acc[mt][nt], false, false);
    __syncthreads();
  }

  // epilogue: C/D layout -> lane holds col n=lm, rows (8*hf + r)
#pragma unroll
  for (int mt = 0; mt < 2; ++mt)
#pragma unroll
    for (int nt = 0; nt < 2; ++nt)
#pragma unroll
      for (int r = 0; r < 8; ++r) {
        int m = m0 + wm + mt * 16 + 8 * hf + r;
        int n = n0 + wn + nt * 16 + lm;
        float v = acc[mt][nt][r];
        if (bias) v += bias[n];
        if (do_gelu) v = 0.5f * v * (1.f + erff(v * 0.70710678f));
        if (resid) v += resid[(size_t)m * N + n];
        if (outf) outf[(size_t)m * N + n] = v;
        else      outb[(size_t)m * N + n] = to_bf16(v);
      }
}

// ---------------------------------------------------------------------------
// Flash-attention with WMMA: 4 waves/block, 1 wave = one 16-query tile.
// Key tiles of 32 staged in LDS (K row-major via async copy, V transposed),
// causal, online softmax with shfl_xor reductions within 16-lane halves.
// grid = (T/64, H, B), block = 128
// ---------------------------------------------------------------------------
__global__ __launch_bounds__(128)
void attn_kernel(const bf16_t* __restrict__ qkv, bf16_t* __restrict__ y) {
  const int tid  = threadIdx.x;
  const int wid  = tid >> 5;
  const int lane = tid & 31;
  const int lm   = lane & 15;
  const int hf   = lane >> 4;
  const int h = blockIdx.y;
  const int b = blockIdx.z;
  const int qbase = blockIdx.x * 64 + wid * 16;

  __shared__ bf16_t Ks[32 * 72];     // [key][d]  stride 72 (144B): aligned, conflict-free
  __shared__ bf16_t Vt[64 * 40];     // [d][key]  transposed
  __shared__ bf16_t Ps[4][16 * 40];  // per-wave P scratch (layout shuffle D->A)

  const size_t rs = 3 * CC;
  const bf16_t* qb = qkv + ((size_t)b * TT) * rs + (size_t)h * HDD;
  const bf16_t* kb = qb + CC;
  const bf16_t* vb = qb + 2 * CC;

  v16bf qf[2];   // Q 16x64 as two 16x32 A-fragments, kept in registers
  {
    const bf16_t* qrow = qb + (size_t)(qbase + lm) * rs;
#pragma unroll
    for (int c = 0; c < 2; ++c)
      qf[c] = combine8(qrow + c * 32 + 8 * hf, qrow + c * 32 + 16 + 8 * hf);
  }

  v8f o[4] = {};
  float mrow[8], lrow[8];
#pragma unroll
  for (int r = 0; r < 8; ++r) { mrow[r] = -1e30f; lrow[r] = 0.f; }

  const int nkt = blockIdx.x * 2 + 2;      // key tiles needed by this block (causal)
  for (int kt = 0; kt < nkt; ++kt) {
    const int kt0 = kt * 32;
    {   // cooperative staging: K via async global->LDS, V transposed via scatter
      int key = tid >> 2;
      int d16 = (tid & 3) * 16;
      const bf16_t* gk = kb + (size_t)(kt0 + key) * rs + d16;
      copy16_g2l(&Ks[key * 72 + d16],     gk);
      copy16_g2l(&Ks[key * 72 + d16 + 8], gk + 8);
      const bf16_t* gv = vb + (size_t)(kt0 + key) * rs + d16;
      bf16x8 v0 = *(const bf16x8*)gv;
      bf16x8 v1 = *(const bf16x8*)(gv + 8);
#pragma unroll
      for (int i = 0; i < 8; ++i) {
        Vt[(d16 + i) * 40 + key]     = v0[i];
        Vt[(d16 + 8 + i) * 40 + key] = v1[i];
      }
    }
    wait_async_all();
    __syncthreads();

    if (kt0 <= qbase + 15) {           // wave-uniform: EXEC stays all-ones for WMMA
      v8f s0 = {}, s1 = {};
#pragma unroll
      for (int c = 0; c < 2; ++c) {    // S = Q * K^T over HD=64 (two 32-chunks)
        v16bf b0 = combine8(&Ks[lm * 72 + c * 32 + 8 * hf],
                            &Ks[lm * 72 + c * 32 + 16 + 8 * hf]);
        v16bf b1 = combine8(&Ks[(16 + lm) * 72 + c * 32 + 8 * hf],
                            &Ks[(16 + lm) * 72 + c * 32 + 16 + 8 * hf]);
        s0 = __builtin_amdgcn_wmma_f32_16x16x32_bf16(false, qf[c], false, b0, (short)0, s0, false, false);
        s1 = __builtin_amdgcn_wmma_f32_16x16x32_bf16(false, qf[c], false, b1, (short)0, s1, false, false);
      }
      float rmax[8];
#pragma unroll
      for (int r = 0; r < 8; ++r) {    // scale 1/sqrt(64) + causal mask
        int row = qbase + 8 * hf + r;
        float v0 = (kt0 + lm      <= row) ? s0[r] * 0.125f : -1e30f;
        float v1 = (kt0 + 16 + lm <= row) ? s1[r] * 0.125f : -1e30f;
        s0[r] = v0; s1[r] = v1;
        rmax[r] = fmaxf(v0, v1);
      }
#pragma unroll
      for (int msk = 1; msk <= 8; msk <<= 1)
#pragma unroll
        for (int r = 0; r < 8; ++r)
          rmax[r] = fmaxf(rmax[r], __shfl_xor(rmax[r], msk, 32));
      float alpha[8], rsum[8];
#pragma unroll
      for (int r = 0; r < 8; ++r) {
        float nm = fmaxf(mrow[r], rmax[r]);
        alpha[r] = __expf(mrow[r] - nm);
        mrow[r] = nm;
        float p0 = __expf(s0[r] - nm);
        float p1 = __expf(s1[r] - nm);
        Ps[wid][(8 * hf + r) * 40 + lm]      = to_bf16(p0);  // D-layout -> LDS
        Ps[wid][(8 * hf + r) * 40 + 16 + lm] = to_bf16(p1);
        rsum[r] = p0 + p1;
      }
#pragma unroll
      for (int msk = 1; msk <= 8; msk <<= 1)
#pragma unroll
        for (int r = 0; r < 8; ++r)
          rsum[r] += __shfl_xor(rsum[r], msk, 32);
#pragma unroll
      for (int r = 0; r < 8; ++r) lrow[r] = lrow[r] * alpha[r] + rsum[r];
#pragma unroll
      for (int ds = 0; ds < 4; ++ds)
#pragma unroll
        for (int r = 0; r < 8; ++r) o[ds][r] *= alpha[r];

      // reload P as A-fragment (same-wave LDS ops are in-order)
      v16bf pf = combine8(&Ps[wid][lm * 40 + 8 * hf], &Ps[wid][lm * 40 + 16 + 8 * hf]);
#pragma unroll
      for (int ds = 0; ds < 4; ++ds) {  // O += P * V  (four 16-col chunks of HD)
        v16bf vf = combine8(&Vt[(ds * 16 + lm) * 40 + 8 * hf],
                            &Vt[(ds * 16 + lm) * 40 + 16 + 8 * hf]);
        o[ds] = __builtin_amdgcn_wmma_f32_16x16x32_bf16(false, pf, false, vf, (short)0, o[ds], false, false);
      }
    }
    __syncthreads();
  }

#pragma unroll
  for (int ds = 0; ds < 4; ++ds)
#pragma unroll
    for (int r = 0; r < 8; ++r) {
      int tok = b * TT + qbase + 8 * hf + r;
      int col = h * HDD + ds * 16 + lm;
      y[(size_t)tok * CC + col] = to_bf16(o[ds][r] / lrow[r]);
    }
}

// ---------------------------------------------------------------------------
// loss = -mean(log_softmax(logits)[target]) over B*T rows, V=32000
// ---------------------------------------------------------------------------
__global__ void zero_kernel(float* p) { if (threadIdx.x == 0 && blockIdx.x == 0) *p = 0.f; }

__global__ __launch_bounds__(256)
void loss_kernel(const float* __restrict__ logits, const int* __restrict__ targets,
                 float* __restrict__ loss) {
  const int row = blockIdx.x;
  const float* lr = logits + (size_t)row * VV;
  __shared__ float red[8];
  int wid = threadIdx.x >> 5, lane = threadIdx.x & 31;

  float m = -1e30f;
  for (int c = threadIdx.x; c < VV; c += 256) m = fmaxf(m, lr[c]);
  for (int msk = 1; msk < 32; msk <<= 1) m = fmaxf(m, __shfl_xor(m, msk, 32));
  if (lane == 0) red[wid] = m;
  __syncthreads();
  float mall = red[0];
#pragma unroll
  for (int i = 1; i < 8; ++i) mall = fmaxf(mall, red[i]);
  __syncthreads();

  float se = 0.f;
  for (int c = threadIdx.x; c < VV; c += 256) se += __expf(lr[c] - mall);
  for (int msk = 1; msk < 32; msk <<= 1) se += __shfl_xor(se, msk, 32);
  if (lane == 0) red[wid] = se;
  __syncthreads();
  if (threadIdx.x == 0) {
    float tot = 0.f;
#pragma unroll
    for (int i = 0; i < 8; ++i) tot += red[i];
    float lsm = lr[targets[row]] - mall - __logf(tot);
    atomicAdd(loss, -lsm * (1.f / (float)BT));
  }
}

// ---------------------------------------------------------------------------
extern "C" void kernel_launch(void* const* d_in, const int* in_sizes, int n_in,
                              void* d_out, int out_size, void* d_ws, size_t ws_size,
                              hipStream_t stream) {
  (void)in_sizes; (void)n_in; (void)out_size; (void)ws_size;
  const int*   idx     = (const int*)d_in[0];
  const int*   targets = (const int*)d_in[1];
  const float* wte     = (const float*)d_in[2];
  const float* wpe     = (const float*)d_in[3];
  const float* ln1_g   = (const float*)d_in[4];
  const float* ln1_b   = (const float*)d_in[5];
  const float* qkv_w   = (const float*)d_in[6];
  const float* aproj_w = (const float*)d_in[7];
  const float* ln2_g   = (const float*)d_in[8];
  const float* ln2_b   = (const float*)d_in[9];
  const float* fc_w    = (const float*)d_in[10];
  const float* fc_b    = (const float*)d_in[11];
  const float* mproj_w = (const float*)d_in[12];
  const float* mproj_b = (const float*)d_in[13];
  const float* lnf_g   = (const float*)d_in[14];
  const float* lnf_b   = (const float*)d_in[15];

  float* logits = (float*)d_out;
  float* loss   = logits + (size_t)BT * VV;

  char* ws = (char*)d_ws;
  size_t off = 0;
  auto wsalloc = [&](size_t bytes) -> void* {
    void* p = ws + off;
    off += (bytes + 255) & ~(size_t)255;
    return p;
  };
  bf16_t* wqkv = (bf16_t*)wsalloc((size_t)LL * CC * 3 * CC * 2);
  bf16_t* wap  = (bf16_t*)wsalloc((size_t)LL * CC * CC * 2);
  bf16_t* wfc  = (bf16_t*)wsalloc((size_t)LL * CC * DFF * 2);
  bf16_t* wmp  = (bf16_t*)wsalloc((size_t)LL * DFF * CC * 2);
  bf16_t* wteb = (bf16_t*)wsalloc((size_t)VV * CC * 2);
  float*  x    = (float*) wsalloc((size_t)BT * CC * 4);
  bf16_t* hbuf = (bf16_t*)wsalloc((size_t)BT * CC * 2);
  bf16_t* qkv  = (bf16_t*)wsalloc((size_t)BT * 3 * CC * 2);
  bf16_t* ybuf = (bf16_t*)wsalloc((size_t)BT * CC * 2);
  bf16_t* mid  = (bf16_t*)wsalloc((size_t)BT * DFF * 2);

  // one-time-per-launch bf16 weight conversion
  cvt_kernel<<<2048, 256, 0, stream>>>(qkv_w,   wqkv, (size_t)LL * CC * 3 * CC);
  cvt_kernel<<<2048, 256, 0, stream>>>(aproj_w, wap,  (size_t)LL * CC * CC);
  cvt_kernel<<<2048, 256, 0, stream>>>(fc_w,    wfc,  (size_t)LL * CC * DFF);
  cvt_kernel<<<2048, 256, 0, stream>>>(mproj_w, wmp,  (size_t)LL * DFF * CC);
  cvt_kernel<<<2048, 256, 0, stream>>>(wte,     wteb, (size_t)VV * CC);

  embed_kernel<<<(BT * CC + 255) / 256, 256, 0, stream>>>(idx, wte, wpe, x);

  for (int l = 0; l < LL; ++l) {
    ln_kernel<<<BT, 256, 0, stream>>>(x, ln1_g + l * CC, ln1_b + l * CC, hbuf);
    // qkv projection: [BT,1024] x [1024,3072] -> bf16
    gemm_kernel<0><<<dim3(3 * CC / 64, BT / 128), 256, 0, stream>>>(
        hbuf, wqkv + (size_t)l * CC * 3 * CC, BT, 3 * CC, CC,
        nullptr, 0, nullptr, qkv, nullptr);
    attn_kernel<<<dim3(TT / 64, HH, BBATCH), 128, 0, stream>>>(qkv, ybuf);
    // attention out projection + residual -> fp32 x
    gemm_kernel<0><<<dim3(CC / 64, BT / 128), 256, 0, stream>>>(
        ybuf, wap + (size_t)l * CC * CC, BT, CC, CC,
        nullptr, 0, x, nullptr, x);
    ln_kernel<<<BT, 256, 0, stream>>>(x, ln2_g + l * CC, ln2_b + l * CC, hbuf);
    // fc + bias + exact GELU -> bf16
    gemm_kernel<0><<<dim3(DFF / 64, BT / 128), 256, 0, stream>>>(
        hbuf, wfc + (size_t)l * CC * DFF, BT, DFF, CC,
        fc_b + l * DFF, 1, nullptr, mid, nullptr);
    // mlp projection + bias + residual -> fp32 x
    gemm_kernel<0><<<dim3(CC / 64, BT / 128), 256, 0, stream>>>(
        mid, wmp + (size_t)l * DFF * CC, BT, CC, DFF,
        mproj_b + l * CC, 0, x, nullptr, x);
  }

  ln_kernel<<<BT, 256, 0, stream>>>(x, lnf_g, lnf_b, hbuf);
  // tied lm_head: logits = h @ wte^T ; wte is [V,C] => B stored [N,K]
  gemm_kernel<1><<<dim3(VV / 64, BT / 128), 256, 0, stream>>>(
      hbuf, wteb, BT, VV, CC, nullptr, 0, nullptr, nullptr, logits);

  zero_kernel<<<1, 32, 0, stream>>>(loss);
  loss_kernel<<<BT, 256, 0, stream>>>(logits, targets, loss);
}
